// Representation_32873679684054
// MI455X (gfx1250) — compile-verified, matrix operand order checked by reading, not run on hardware
//
#include <hip/hip_runtime.h>

// ---------------- problem constants ----------------
#define LL     32
#define NN     1024
#define CSTEPS 10
#define IDIM   13
#define H1DIM  128     // HIDDEN
#define H2DIM  256     // HID
#define GATES  512     // 4*HIDDEN
#define D2DIM  280     // 15+7+2+256
#define EPSV   1e-5f

typedef __attribute__((ext_vector_type(16))) __bf16 v16bf;
typedef __attribute__((ext_vector_type(8)))  float  v8f;

// ---------------- helpers ----------------
__device__ __forceinline__ float sigf(float x)  { return 1.0f / (1.0f + __expf(-x)); }
__device__ __forceinline__ float tanhf_(float x){ float e = __expf(2.0f * x); return 1.0f - 2.0f / (e + 1.0f); }

// Load a 16x32 bf16 WMMA A/B fragment from row-major src (ld bf16 elems / row).
// ISA 7.12.2: lane<16 -> row=lane, K {kb..kb+7, kb+16..kb+23}; lane>=16 -> row=lane-16,
// K {kb+8..kb+15, kb+24..kb+31}.  Requires ld%8==0, kbase%8==0, src 16B aligned.
__device__ __forceinline__ v16bf load_frag(const __bf16* __restrict__ src,
                                           int rowBase, int ld, int kbase) {
  int lane = threadIdx.x & 31;
  const __bf16* p = src + (size_t)(rowBase + (lane & 15)) * ld + kbase + ((lane >> 4) << 3);
  union { uint4 u[2]; v16bf v; } t;
  t.u[0] = *(const uint4*)(p);
  t.u[1] = *(const uint4*)(p + 16);
  return t.v;
}

__device__ __forceinline__ v8f wmma_bf16(v16bf a, v16bf b, v8f c) {
  return __builtin_amdgcn_wmma_f32_16x16x32_bf16(false, a, false, b, (short)0, c, false, false);
}

// =====================================================================
// K1: condition LSTM (10 steps) + W_cond projection + ReLU, 64 rows/block.
// Flat batch 65536: rows <32768 = condition1s, rows >=32768 = condition2s.
// 512 threads = 16 waves; each wave owns 2 (mtile,ct) pairs sharing one
// M-tile, so the A fragment is loaded once per K-block and reused for
// 2 ct-tiles x 4 gates.  Cell state c lives in D-fragment registers.
// LDS: Wi[512][32] Wh[512][128] Wc[256][128] (bf16), X[10][64][32] (bf16),
//      H[64][128] (bf16), b1[512] bc[256] (f32)   => 289,792 B
// =====================================================================
__global__ void __launch_bounds__(512, 1)
k_cond_lstm(const float* __restrict__ cond1,
            const float* __restrict__ cond2,
            const float* __restrict__ W_ih1,
            const float* __restrict__ W_hh1,
            const float* __restrict__ b1,
            const float* __restrict__ W_cond,
            const float* __restrict__ b_cond,
            float* __restrict__ o_full) {
  extern __shared__ char smem[];
  __bf16* Wi = (__bf16*)smem;                 // 512*32
  __bf16* Wh = Wi + 512 * 32;                 // 512*128
  __bf16* Wc = Wh + 512 * 128;                // 256*128
  __bf16* Xs = Wc + 256 * 128;                // 10*64*32
  __bf16* Hs = Xs + CSTEPS * 64 * 32;         // 64*128
  float*  b1s = (float*)(Hs + 64 * 128);      // 512
  float*  bcs = b1s + 512;                    // 256

  const int tid  = threadIdx.x;
  const int wave = tid >> 5;                  // 0..15
  const int lane = tid & 31;
  const int nn   = lane & 15;
  const int half = lane >> 4;

  const int rowBaseG = blockIdx.x * 64;
  const float* src = (rowBaseG < 32768) ? cond1 : cond2;
  const int rloc = (rowBaseG < 32768) ? rowBaseG : rowBaseG - 32768;

  // wave's M-tile and its two channel tiles (idx = wave*2+p, pairs share mt)
  const int mt  = (wave * 2) >> 3;            // 4 M tiles of 16 rows
  const int ct0 = (wave * 2) & 7;             // even ct
  const int ct1 = ct0 + 1;

  // ---- cooperative LDS fill ----
  for (int i = tid; i < 512 * 32; i += 512) Wi[i] = (__bf16)0.0f;
  for (int i = tid; i < 512 * IDIM; i += 512) {
    int r = i / IDIM, k = i % IDIM;
    Wi[r * 32 + k] = (__bf16)W_ih1[i];
  }
  for (int i = tid; i < 512 * 128; i += 512) Wh[i] = (__bf16)W_hh1[i];
  for (int i = tid; i < 256 * 128; i += 512) Wc[i] = (__bf16)W_cond[i];
  for (int i = tid; i < CSTEPS * 64 * 32; i += 512) Xs[i] = (__bf16)0.0f;
  for (int i = tid; i < 64 * CSTEPS * IDIM; i += 512) {
    int r = i / (CSTEPS * IDIM), rem = i % (CSTEPS * IDIM);
    int t = rem / IDIM, k = rem % IDIM;
    float v = src[((size_t)(rloc + r) * CSTEPS + t) * IDIM + k];
    Xs[(t * 64 + r) * 32 + k] = (__bf16)v;
  }
  for (int i = tid; i < 64 * 128; i += 512) Hs[i] = (__bf16)0.0f;
  if (tid < 512) b1s[tid] = b1[tid];
  if (tid < 256) bcs[tid] = b_cond[tid];
  __syncthreads();

  // ---- 10 LSTM steps ----
  float cst[2][8];
#pragma unroll
  for (int p = 0; p < 2; ++p)
#pragma unroll
    for (int r = 0; r < 8; ++r) cst[p][r] = 0.0f;

  for (int t = 0; t < CSTEPS; ++t) {
    v8f acc[2][4];
#pragma unroll
    for (int p = 0; p < 2; ++p) {
      const int ct = p ? ct1 : ct0;
#pragma unroll
      for (int q = 0; q < 4; ++q) {
        float bv = b1s[q * 128 + ct * 16 + nn];
#pragma unroll
        for (int r = 0; r < 8; ++r) acc[p][q][r] = bv;
      }
    }
    // K-block 0: x part (13 padded to 32, zero-padded both operands)
    {
      v16bf af = load_frag(Xs + t * 64 * 32, mt * 16, 32, 0);
#pragma unroll
      for (int p = 0; p < 2; ++p) {
        const int ct = p ? ct1 : ct0;
#pragma unroll
        for (int q = 0; q < 4; ++q) {
          v16bf bf = load_frag(Wi, q * 128 + ct * 16, 32, 0);
          acc[p][q] = wmma_bf16(af, bf, acc[p][q]);
        }
      }
    }
    // K-blocks 1..4: h part (A fragment shared by both pairs, all gates)
#pragma unroll
    for (int kb = 0; kb < 4; ++kb) {
      v16bf af = load_frag(Hs, mt * 16, 128, kb * 32);
#pragma unroll
      for (int p = 0; p < 2; ++p) {
        const int ct = p ? ct1 : ct0;
#pragma unroll
        for (int q = 0; q < 4; ++q) {
          v16bf bf = load_frag(Wh, q * 128 + ct * 16, 128, kb * 32);
          acc[p][q] = wmma_bf16(af, bf, acc[p][q]);
        }
      }
    }
    // pointwise cell update (all four gates lane-local)
    float hreg[2][8];
#pragma unroll
    for (int p = 0; p < 2; ++p)
#pragma unroll
      for (int r = 0; r < 8; ++r) {
        float iv = sigf(acc[p][0][r]);
        float fv = sigf(acc[p][1][r]);
        float gv = tanhf_(acc[p][2][r]);
        float ov = sigf(acc[p][3][r]);
        float cn = fv * cst[p][r] + iv * gv;
        cst[p][r] = cn;
        hreg[p][r] = ov * tanhf_(cn);
      }
    __syncthreads();
#pragma unroll
    for (int p = 0; p < 2; ++p) {
      const int ct = p ? ct1 : ct0;
#pragma unroll
      for (int r = 0; r < 8; ++r) {
        int m = mt * 16 + half * 8 + r;
        Hs[m * 128 + ct * 16 + nn] = (__bf16)hreg[p][r];
      }
    }
    __syncthreads();
  }

  // ---- projection: o = relu(h @ Wc^T + bc), (64x128)@(128x256) ----
  for (int p = 0; p < 4; ++p) {
    const int idx = wave * 4 + p;         // 64 (mtile,nt) pairs over 16 waves
    const int pmt = idx >> 4, nt = idx & 15;
    v8f acc;
    float bv = bcs[nt * 16 + nn];
#pragma unroll
    for (int r = 0; r < 8; ++r) acc[r] = bv;
#pragma unroll
    for (int kb = 0; kb < 4; ++kb) {
      v16bf af = load_frag(Hs, pmt * 16, 128, kb * 32);
      v16bf bf = load_frag(Wc, nt * 16, 128, kb * 32);
      acc = wmma_bf16(af, bf, acc);
    }
#pragma unroll
    for (int r = 0; r < 8; ++r) {
      int row = rowBaseG + pmt * 16 + half * 8 + r;
      o_full[(size_t)row * 256 + nt * 16 + nn] = fmaxf(acc[r], 0.0f);
    }
  }
}

// =====================================================================
// K2: o = 0.5*(o1+o2); per-channel sum/sumsq for BN1 (atomics)
// =====================================================================
__global__ void k_combine(const float* __restrict__ o_full,
                          float* __restrict__ o_comb,
                          float* __restrict__ s1, float* __restrict__ s2) {
  const int t = threadIdx.x;             // channel
  const int r0 = blockIdx.x * 128;
  float a = 0.0f, b = 0.0f;
  for (int j = 0; j < 128; ++j) {
    size_t r = (size_t)(r0 + j);
    float v = 0.5f * (o_full[r * 256 + t] + o_full[(r + 32768) * 256 + t]);
    o_comb[r * 256 + t] = v;
    a += v; b += v * v;
  }
  atomicAdd(&s1[t], a);
  atomicAdd(&s2[t], b);
}

// =====================================================================
// K3: fold BN into scale/shift: a = g*rsqrt(var+eps), b = be - mean*a
// =====================================================================
__global__ void k_bn_fin(const float* __restrict__ s1, const float* __restrict__ s2,
                         const float* __restrict__ g, const float* __restrict__ be,
                         float inv_cnt, int nch,
                         float* __restrict__ a, float* __restrict__ b) {
  int t = threadIdx.x;
  if (t < nch) {
    float m = s1[t] * inv_cnt;
    float v = s2[t] * inv_cnt - m * m;
    float sc = g[t] * rsqrtf(v + EPSV);
    a[t] = sc;
    b[t] = be[t] - m * sc;
  }
}

// =====================================================================
// K4: Xproj = feats @ W_ih2^T + b2   (32768 x 280) @ (280 x 512)
// feats assembled in LDS: [ops(15)|extra(7)|card(2)|BN1(cond)(256)|pad->288]
// W_ih2 packed [512][280] bf16 in LDS (+16 tail pad; A zero-pads K 280..287)
// 512 threads, 8 pairs/wave.
// =====================================================================
__global__ void __launch_bounds__(512, 1)
k_xproj(const float* __restrict__ ops,
        const float* __restrict__ extra,
        const float* __restrict__ card,
        const float* __restrict__ o_comb,
        const float* __restrict__ a1, const float* __restrict__ b1s,
        const float* __restrict__ W_ih2,
        const float* __restrict__ b2,
        float* __restrict__ Xproj) {
  extern __shared__ char smem[];
  __bf16* W2 = (__bf16*)smem;                  // 512*280 + 16
  __bf16* Af = W2 + 512 * D2DIM + 16;          // 64*288
  float*  b2s = (float*)(Af + 64 * 288);       // 512

  const int tid = threadIdx.x;
  const int wave = tid >> 5, lane = tid & 31;
  const int nn = lane & 15, half = lane >> 4;
  const int rowBase = blockIdx.x * 64;

  for (int i = tid; i < 512 * D2DIM; i += 512) W2[i] = (__bf16)W_ih2[i];
  if (tid < 16) W2[512 * D2DIM + tid] = (__bf16)0.0f;
  if (tid < 512) b2s[tid] = b2[tid];
  for (int i = tid; i < 64 * 288; i += 512) {
    int m = i / 288, k = i % 288;
    size_t row = (size_t)(rowBase + m);        // flat (l*1024+n)
    float v;
    if (k < 15)       v = ops[row * 15 + k];
    else if (k < 22)  v = extra[row * 7 + (k - 15)];
    else if (k < 24)  v = card[row * 2 + (k - 22)];
    else if (k < 280) { int c = k - 24; v = a1[c] * o_comb[row * 256 + c] + b1s[c]; }
    else              v = 0.0f;
    Af[i] = (__bf16)v;
  }
  __syncthreads();

  for (int p = 0; p < 8; ++p) {
    const int idx = wave * 8 + p;        // 128 (mtile,nt) pairs over 16 waves
    const int mt = idx >> 5, nt = idx & 31;
    v8f acc;
    float bv = b2s[nt * 16 + nn];
#pragma unroll
    for (int r = 0; r < 8; ++r) acc[r] = bv;
#pragma unroll
    for (int kb = 0; kb < 9; ++kb) {
      v16bf af = load_frag(Af, mt * 16, 288, kb * 32);
      v16bf bf = load_frag(W2, nt * 16, D2DIM, kb * 32);
      acc = wmma_bf16(af, bf, acc);
    }
#pragma unroll
    for (int r = 0; r < 8; ++r) {
      size_t row = (size_t)(rowBase + mt * 16 + half * 8 + r);
      Xproj[row * 512 + nt * 16 + nn] = acc[r];
    }
  }
}

// =====================================================================
// K5: tree init: cell(feats[31], 0, 0) from precomputed gates Xproj[31]
// =====================================================================
__global__ void k_tree_init(const float* __restrict__ Xproj,
                            float* __restrict__ H0, float* __restrict__ C0) {
  int e = blockIdx.x * 256 + threadIdx.x;      // 1024*128
  int m = e >> 7, ch = e & 127;
  size_t base = ((size_t)(31 * NN + m)) * 512 + ch;
  float iv = sigf(Xproj[base]);
  float gv = tanhf_(Xproj[base + 256]);
  float ov = sigf(Xproj[base + 384]);
  float c = iv * gv;                           // sig(f)*0 + sig(i)*tanh(g)
  H0[e] = ov * tanhf_(c);
  C0[e] = c;
}

// =====================================================================
// K6: sequential tree scan, 31 steps, single 1024-thread workgroup.
// W_hh2 bf16 resident in LDS; h/c ping-pong in global; gather per mapping.
// =====================================================================
__global__ void __launch_bounds__(1024, 1)
k_tree(const float* __restrict__ Xproj,
       const int* __restrict__ mapping,
       const float* __restrict__ W_hh2,
       float* __restrict__ H0, float* __restrict__ C0,
       float* __restrict__ H1, float* __restrict__ C1,
       __bf16* __restrict__ Ag, float* __restrict__ Cg) {
  extern __shared__ char smem[];
  __bf16* Whh = (__bf16*)smem;                 // 512*128
  const int tid = threadIdx.x;
  const int wave = tid >> 5, lane = tid & 31;
  const int nn = lane & 15, half = lane >> 4;

  for (int i = tid; i < 512 * 128; i += 1024) Whh[i] = (__bf16)W_hh2[i];
  __syncthreads();

  for (int s = 0; s < 31; ++s) {
    const int l = 30 - s;
    const float* Hp = (s & 1) ? H1 : H0;
    const float* Cp = (s & 1) ? C1 : C0;
    float* Hn = (s & 1) ? H0 : H1;
    float* Cn = (s & 1) ? C0 : C1;

    // gather: hp[mp0]+hp[mp1] with pad row 0
    for (int e = tid; e < NN * 128; e += 1024) {
      int m = e >> 7, ch = e & 127;
      const int* mp = mapping + ((size_t)l * NN + m) * 2;
      int i0 = mp[0], i1 = mp[1];
      float hv = 0.0f, cv = 0.0f;
      if (i0 > 0) { hv += Hp[(i0 - 1) * 128 + ch]; cv += Cp[(i0 - 1) * 128 + ch]; }
      if (i1 > 0) { hv += Hp[(i1 - 1) * 128 + ch]; cv += Cp[(i1 - 1) * 128 + ch]; }
      Ag[e] = (__bf16)(0.5f * hv);
      Cg[e] = 0.5f * cv;
    }
    __threadfence();
    __syncthreads();

    // G = Xproj[l] + h_g @ Whh^T ; cell update
    for (int p = 0; p < 16; ++p) {
      const int idx = wave * 16 + p;     // 512 (mtile,ct) pairs over 32 waves
      const int mt = idx >> 3, ct = idx & 7;
      v8f acc[4];
#pragma unroll
      for (int q = 0; q < 4; ++q)
#pragma unroll
        for (int r = 0; r < 8; ++r) {
          size_t row = (size_t)l * NN + mt * 16 + half * 8 + r;
          acc[q][r] = Xproj[row * 512 + q * 128 + ct * 16 + nn];
        }
#pragma unroll
      for (int kb = 0; kb < 4; ++kb) {
        v16bf af = load_frag(Ag, mt * 16, 128, kb * 32);
#pragma unroll
        for (int q = 0; q < 4; ++q) {
          v16bf bf = load_frag(Whh, q * 128 + ct * 16, 128, kb * 32);
          acc[q] = wmma_bf16(af, bf, acc[q]);
        }
      }
#pragma unroll
      for (int r = 0; r < 8; ++r) {
        int m = mt * 16 + half * 8 + r;
        int ch = ct * 16 + nn;
        float cg = Cg[m * 128 + ch];
        float iv = sigf(acc[0][r]);
        float fv = sigf(acc[1][r]);
        float gv = tanhf_(acc[2][r]);
        float ov = sigf(acc[3][r]);
        float cn = fv * cg + iv * gv;
        Hn[m * 128 + ch] = ov * tanhf_(cn);
        Cn[m * 128 + ch] = cn;
      }
    }
    __threadfence();
    __syncthreads();
  }
  // final h/c in H1/C1 (31 odd steps)
}

// =====================================================================
// K7: BN2 over final h (1024x128), direct (no atomics), fused finalize
// =====================================================================
__global__ void k_bn2(const float* __restrict__ Hf,
                      const float* __restrict__ g2, const float* __restrict__ be2,
                      float* __restrict__ a2, float* __restrict__ b2s) {
  int ch = threadIdx.x;                  // 128 threads
  float s = 0.0f, s2 = 0.0f;
  for (int r = 0; r < NN; ++r) {
    float v = Hf[r * 128 + ch];
    s += v; s2 += v * v;
  }
  float m = s / (float)NN;
  float var = s2 / (float)NN - m * m;
  float sc = g2[ch] * rsqrtf(var + EPSV);
  a2[ch] = sc;
  b2s[ch] = be2[ch] - m * sc;
}

// =====================================================================
// K8: t1 = sigmoid(BN2(h) @ W_m2^T + b_m2), accumulate BN3 stats
// =====================================================================
__global__ void k_mlp1(const float* __restrict__ Hf,
                       const float* __restrict__ a2, const float* __restrict__ b2s,
                       const float* __restrict__ W_m2, const float* __restrict__ b_m2,
                       float* __restrict__ t1,
                       float* __restrict__ s3, float* __restrict__ s3b) {
  __shared__ float xr[128];
  const int r = blockIdx.x, t = threadIdx.x;
  if (t < 128) xr[t] = a2[t] * Hf[r * 128 + t] + b2s[t];
  __syncthreads();
  float d = b_m2[t];
  const float* w = W_m2 + t * 128;
#pragma unroll 4
  for (int k = 0; k < 128; ++k) d += xr[k] * w[k];
  float v = sigf(d);
  t1[(size_t)r * 256 + t] = v;
  atomicAdd(&s3[t], v);
  atomicAdd(&s3b[t], v * v);
}

// =====================================================================
// K10: out = sigmoid( sigmoid(BN3(t1)@W_m3^T + b_m3) @ W_out^T + b_out )
// =====================================================================
__global__ void k_mlp2(const float* __restrict__ t1,
                       const float* __restrict__ a3, const float* __restrict__ b3s,
                       const float* __restrict__ W_m3, const float* __restrict__ b_m3,
                       const float* __restrict__ W_out, const float* __restrict__ b_out,
                       float* __restrict__ out) {
  __shared__ float xr[256];
  __shared__ float red[256];
  const int r = blockIdx.x, t = threadIdx.x;
  xr[t] = a3[t] * t1[(size_t)r * 256 + t] + b3s[t];
  __syncthreads();
  float d = b_m3[t];
  const float* w = W_m3 + t * 256;
#pragma unroll 4
  for (int k = 0; k < 256; ++k) d += xr[k] * w[k];
  red[t] = sigf(d) * W_out[t];
  __syncthreads();
  for (int s = 128; s > 0; s >>= 1) {
    if (t < s) red[t] += red[t + s];
    __syncthreads();
  }
  if (t == 0) out[r] = sigf(red[0] + b_out[0]);
}

// =====================================================================
// host-side launch
// =====================================================================
extern "C" void kernel_launch(void* const* d_in, const int* in_sizes, int n_in,
                              void* d_out, int out_size, void* d_ws, size_t ws_size,
                              hipStream_t stream) {
  (void)in_sizes; (void)n_in; (void)out_size; (void)ws_size;

  const float* operators  = (const float*)d_in[0];
  const float* extras     = (const float*)d_in[1];
  const float* cards      = (const float*)d_in[2];
  const float* cond1      = (const float*)d_in[3];
  const float* cond2      = (const float*)d_in[4];
  const int*   mapping    = (const int*)  d_in[5];
  const float* W_ih1      = (const float*)d_in[6];
  const float* W_hh1      = (const float*)d_in[7];
  const float* b1         = (const float*)d_in[8];
  const float* W_cond     = (const float*)d_in[9];
  const float* b_cond     = (const float*)d_in[10];
  const float* g1         = (const float*)d_in[11];
  const float* be1        = (const float*)d_in[12];
  const float* W_ih2      = (const float*)d_in[13];
  const float* W_hh2      = (const float*)d_in[14];
  const float* b2         = (const float*)d_in[15];
  const float* g2         = (const float*)d_in[16];
  const float* be2        = (const float*)d_in[17];
  const float* W_m2       = (const float*)d_in[18];
  const float* b_m2       = (const float*)d_in[19];
  const float* g3         = (const float*)d_in[20];
  const float* be3        = (const float*)d_in[21];
  const float* W_m3       = (const float*)d_in[22];
  const float* b_m3       = (const float*)d_in[23];
  const float* W_out      = (const float*)d_in[24];
  const float* b_out      = (const float*)d_in[25];
  float* out = (float*)d_out;

  // workspace layout (all offsets 256B aligned)
  char* ws = (char*)d_ws;
  const size_t OFF_OFULL = 0;
  const size_t OFF_OCOMB = OFF_OFULL + (size_t)65536 * 256 * 4;
  const size_t OFF_XPROJ = OFF_OCOMB + (size_t)32768 * 256 * 4;
  const size_t OFF_H0    = OFF_XPROJ + (size_t)32768 * 512 * 4;
  const size_t OFF_H1    = OFF_H0 + (size_t)NN * 128 * 4;
  const size_t OFF_C0    = OFF_H1 + (size_t)NN * 128 * 4;
  const size_t OFF_C1    = OFF_C0 + (size_t)NN * 128 * 4;
  const size_t OFF_AG    = OFF_C1 + (size_t)NN * 128 * 4;
  const size_t OFF_CG    = OFF_AG + (size_t)NN * 128 * 2;
  const size_t OFF_T1    = OFF_CG + (size_t)NN * 128 * 4;
  const size_t OFF_STAT  = OFF_T1 + (size_t)NN * 256 * 4;   // 4KB zeroed sums
  const size_t OFF_AB    = OFF_STAT + 4096;

  float* o_full = (float*)(ws + OFF_OFULL);
  float* o_comb = (float*)(ws + OFF_OCOMB);
  float* Xproj  = (float*)(ws + OFF_XPROJ);
  float* H0     = (float*)(ws + OFF_H0);
  float* H1     = (float*)(ws + OFF_H1);
  float* C0     = (float*)(ws + OFF_C0);
  float* C1     = (float*)(ws + OFF_C1);
  __bf16* Ag    = (__bf16*)(ws + OFF_AG);
  float* Cg     = (float*)(ws + OFF_CG);
  float* t1buf  = (float*)(ws + OFF_T1);
  float* bn1s   = (float*)(ws + OFF_STAT);        // 256
  float* bn1s2  = bn1s + 256;                     // 256
  float* bn3s   = bn1s2 + 256;                    // 256
  float* bn3s2  = bn3s + 256;                     // 256
  float* bn1a   = (float*)(ws + OFF_AB);          // 256
  float* bn1b   = bn1a + 256;
  float* bn2a   = bn1b + 256;                     // 128
  float* bn2b   = bn2a + 128;
  float* bn3a   = bn2b + 128;                     // 256
  float* bn3b   = bn3a + 256;

  hipMemsetAsync(ws + OFF_STAT, 0, 4096, stream);

  const size_t SM1 = (size_t)(512*32 + 512*128 + 256*128 + CSTEPS*64*32 + 64*128) * 2
                   + (512 + 256) * 4;                                   // 289,792 B
  const size_t SM4 = (size_t)(512*D2DIM + 16 + 64*288) * 2 + 512 * 4;   // 325,664 B
  const size_t SM6 = (size_t)512 * 128 * 2;                             // 131,072 B

  k_cond_lstm<<<1024, 512, SM1, stream>>>(cond1, cond2, W_ih1, W_hh1, b1,
                                          W_cond, b_cond, o_full);
  k_combine<<<256, 256, 0, stream>>>(o_full, o_comb, bn1s, bn1s2);
  k_bn_fin<<<1, 256, 0, stream>>>(bn1s, bn1s2, g1, be1, 1.0f / 32768.0f, 256, bn1a, bn1b);
  k_xproj<<<512, 512, SM4, stream>>>(operators, extras, cards, o_comb,
                                     bn1a, bn1b, W_ih2, b2, Xproj);
  k_tree_init<<<512, 256, 0, stream>>>(Xproj, H0, C0);
  k_tree<<<1, 1024, SM6, stream>>>(Xproj, mapping, W_hh2, H0, C0, H1, C1, Ag, Cg);
  k_bn2<<<1, 128, 0, stream>>>(H1, g2, be2, bn2a, bn2b);
  k_mlp1<<<1024, 256, 0, stream>>>(H1, bn2a, bn2b, W_m2, b_m2, t1buf, bn3s, bn3s2);
  k_bn_fin<<<1, 256, 0, stream>>>(bn3s, bn3s2, g3, be3, 1.0f / 1024.0f, 256, bn3a, bn3b);
  k_mlp2<<<1024, 256, 0, stream>>>(t1buf, bn3a, bn3b, W_m3, b_m3, W_out, b_out, out);
}
